// LayerNormLSTMCell_31679678775450
// MI455X (gfx1250) — matrix operand order repaired and checked by reading, compiled, not run on hardware
//
#include <hip/hip_runtime.h>
#include <hip/hip_bf16.h>

// ---------------------------------------------------------------------------
// LayerNorm LSTM cell for MI455X (gfx1250, wave32, WMMA).
// Kernel 1: bf16 WMMA GEMM (i2h and h2h via blockIdx.z), f32 accumulate.
//           Block tile 128x256, BK=32, 8 waves, 32x128 per wave (16 WMMA/iter).
//           Fill phase: batched global loads -> staging regs -> cvt -> LDS.
// Kernel 2: fused LayerNorm + gates + cell update + output LayerNorm.
// ---------------------------------------------------------------------------

typedef __attribute__((ext_vector_type(16))) __bf16 v16bf;
typedef __attribute__((ext_vector_type(8)))  float  v8f;
typedef __attribute__((ext_vector_type(4)))  __bf16 v4bf;
typedef __attribute__((ext_vector_type(4)))  unsigned int v4u;

#define MDIM 4096   // batch B
#define NDIM 4096   // 4*H
#define KDIM 1024   // D == H
#define HDIM 1024
#define BM 128
#define BN 256
#define BK 32

union Frag { v16bf v; v4u q[2]; };

__global__ __launch_bounds__(256)
void ln_lstm_gemm(const float* __restrict__ X,  const float* __restrict__ Wih,
                  const float* __restrict__ Hx, const float* __restrict__ Whh,
                  const float* __restrict__ bhh,
                  float* __restrict__ outI, float* __restrict__ outH)
{
    const int z = blockIdx.z;
    const float* __restrict__ A = z ? Hx  : X;
    const float* __restrict__ W = z ? Whh : Wih;
    float* __restrict__ out     = z ? outH : outI;

    __shared__ __bf16 lA[BM][BK];   // [m][k]                      8 KB
    __shared__ __bf16 lB[BN][BK];   // [n][k] (W row-major [N][K]) 16 KB

    const int t       = threadIdx.x;
    const int lane    = t & 31;
    const int wave    = t >> 5;
    const int wm      = wave & 3;     // 4 slabs of 32 rows  -> 128
    const int wn      = wave >> 2;    // 2 slabs of 128 cols -> 256
    const int m0      = blockIdx.y * BM;
    const int n0      = blockIdx.x * BN;
    const int halfsel = lane >> 4;    // 0: lanes 0-15, 1: lanes 16-31
    const int l15     = lane & 15;

    // Per-thread fill coordinates (4 float4 of A tile, 8 float4 of B tile).
    int arow[4], acol[4], brow[8], bcol[8];
    #pragma unroll
    for (int r = 0; r < 4; ++r) {
        const int p = t + 256 * r;
        arow[r] = p >> 3;             // 8 float4 per 32-wide row
        acol[r] = (p & 7) << 2;
    }
    #pragma unroll
    for (int r = 0; r < 8; ++r) {
        const int p = t + 256 * r;
        brow[r] = p >> 3;
        bcol[r] = (p & 7) << 2;
    }

    v8f acc[2][8] = {};

    for (int kk = 0; kk < KDIM; kk += BK) {
        // ---- phase 1: issue ALL global loads back-to-back -----------------
        float4 fa[4], fw[8];
        #pragma unroll
        for (int r = 0; r < 4; ++r)
            fa[r] = *reinterpret_cast<const float4*>(&A[(size_t)(m0 + arow[r]) * KDIM + kk + acol[r]]);
        #pragma unroll
        for (int r = 0; r < 8; ++r)
            fw[r] = *reinterpret_cast<const float4*>(&W[(size_t)(n0 + brow[r]) * KDIM + kk + bcol[r]]);

        // ---- phase 2: one uniform-guarded prefetch block ------------------
        if (kk + BK < KDIM) {
            #pragma unroll
            for (int r = 0; r < 4; ++r)
                __builtin_prefetch(&A[(size_t)(m0 + arow[r]) * KDIM + kk + BK + acol[r]], 0, 3);
            #pragma unroll
            for (int r = 0; r < 8; ++r)
                __builtin_prefetch(&W[(size_t)(n0 + brow[r]) * KDIM + kk + BK + bcol[r]], 0, 3);
        }

        // ---- phase 3: convert f32 -> bf16 and store to LDS ----------------
        #pragma unroll
        for (int r = 0; r < 4; ++r) {
            v4bf pa = { (__bf16)fa[r].x, (__bf16)fa[r].y, (__bf16)fa[r].z, (__bf16)fa[r].w };
            *reinterpret_cast<v4bf*>(&lA[arow[r]][acol[r]]) = pa;
        }
        #pragma unroll
        for (int r = 0; r < 8; ++r) {
            v4bf pw = { (__bf16)fw[r].x, (__bf16)fw[r].y, (__bf16)fw[r].z, (__bf16)fw[r].w };
            *reinterpret_cast<v4bf*>(&lB[brow[r]][bcol[r]]) = pw;
        }
        __syncthreads();

        // ---- build fragments per documented CDNA5 VGPR layouts ------------
        // A 16x32 bf16: lanes 0-15 row=M, K 0-7 then 16-23; lanes 16-31 K 8-15 then 24-31.
        Frag a[2];
        #pragma unroll
        for (int mt = 0; mt < 2; ++mt) {
            const int row = wm * 32 + mt * 16 + l15;
            const int kb  = halfsel * 8;
            a[mt].q[0] = *reinterpret_cast<const v4u*>(&lA[row][kb]);
            a[mt].q[1] = *reinterpret_cast<const v4u*>(&lA[row][kb + 16]);
        }
        // B 32x16 bf16: lanes 0-15 col=N, K 0-7 then 16-23; lanes 16-31 K 8-15 then 24-31.
        Frag b[8];
        #pragma unroll
        for (int nt = 0; nt < 8; ++nt) {
            const int col = wn * 128 + nt * 16 + l15;
            const int kb  = halfsel * 8;
            b[nt].q[0] = *reinterpret_cast<const v4u*>(&lB[col][kb]);
            b[nt].q[1] = *reinterpret_cast<const v4u*>(&lB[col][kb + 16]);
        }

        #pragma unroll
        for (int mt = 0; mt < 2; ++mt)
            #pragma unroll
            for (int nt = 0; nt < 8; ++nt)
                acc[mt][nt] = __builtin_amdgcn_wmma_f32_16x16x32_bf16(
                    false, a[mt].v, false, b[nt].v, (short)0, acc[mt][nt], false, false);

        __syncthreads();
    }

    // ---- epilogue: C/D layout -> global (fuse b_hh for h2h) ---------------
    #pragma unroll
    for (int nt = 0; nt < 8; ++nt) {
        const int col    = n0 + wn * 128 + nt * 16 + l15;
        const float bias = z ? bhh[col] : 0.0f;
        #pragma unroll
        for (int mt = 0; mt < 2; ++mt) {
            const int rbase = m0 + wm * 32 + mt * 16 + halfsel * 8;  // VGPR r -> M = r (+8 hi lanes)
            #pragma unroll
            for (int r = 0; r < 8; ++r)
                out[(size_t)(rbase + r) * NDIM + col] = acc[mt][nt][r] + bias;
        }
    }
}

// ---------------------------------------------------------------------------

__device__ __forceinline__ float sigmoidf_fast(float x) {
    return 1.0f / (1.0f + __expf(-x));
}

__device__ __forceinline__ float4 block_reduce4(float4 v, float4* red, int t) {
    red[t] = v;
    __syncthreads();
    for (int s = 128; s > 0; s >>= 1) {
        if (t < s) {
            float4 o = red[t + s];
            red[t].x += o.x; red[t].y += o.y; red[t].z += o.z; red[t].w += o.w;
        }
        __syncthreads();
    }
    float4 res = red[0];
    __syncthreads();
    return res;
}

__global__ __launch_bounds__(256)
void ln_lstm_pointwise(const float* __restrict__ i2h, const float* __restrict__ h2h,
                       const float* __restrict__ cx,
                       const float* __restrict__ g_ih,  const float* __restrict__ be_ih,
                       const float* __restrict__ g_hh,  const float* __restrict__ be_hh,
                       const float* __restrict__ g_ho,  const float* __restrict__ be_ho,
                       float* __restrict__ hy, float* __restrict__ cy)
{
    const int b = blockIdx.x;
    const int t = threadIdx.x;

    __shared__ float  sI[NDIM];      // i2h row, later reused for normalized gates
    __shared__ float  sH[NDIM];      // h2h row
    __shared__ float4 red[256];

    const float* __restrict__ rowI = i2h + (size_t)b * NDIM;
    const float* __restrict__ rowH = h2h + (size_t)b * NDIM;

    // ---- stage rows + partial sums for both LayerNorm stats ---------------
    float sumI = 0.f, sqI = 0.f, sumH = 0.f, sqH = 0.f;
    #pragma unroll
    for (int r = 0; r < 4; ++r) {
        const int p = (t + 256 * r) << 2;
        float4 a = *reinterpret_cast<const float4*>(&rowI[p]);
        float4 h = *reinterpret_cast<const float4*>(&rowH[p]);
        *reinterpret_cast<float4*>(&sI[p]) = a;
        *reinterpret_cast<float4*>(&sH[p]) = h;
        sumI += a.x + a.y + a.z + a.w;
        sqI  += a.x*a.x + a.y*a.y + a.z*a.z + a.w*a.w;
        sumH += h.x + h.y + h.z + h.w;
        sqH  += h.x*h.x + h.y*h.y + h.z*h.z + h.w*h.w;
    }
    float4 tot = block_reduce4(make_float4(sumI, sqI, sumH, sqH), red, t);
    const float inv = 1.0f / (float)NDIM;
    const float mI = tot.x * inv, vI = tot.y * inv - mI * mI, rI = rsqrtf(vI + 1e-5f);
    const float mH = tot.z * inv, vH = tot.w * inv - mH * mH, rH = rsqrtf(vH + 1e-5f);

    // ---- gates = LN(i2h)*g+b + LN(h2h)*g+b  (in place into sI) ------------
    #pragma unroll
    for (int r = 0; r < 16; ++r) {
        const int j = t + (r << 8);
        const float gv = (sI[j] - mI) * rI * g_ih[j] + be_ih[j]
                       + (sH[j] - mH) * rH * g_hh[j] + be_hh[j];
        sI[j] = gv;                      // each j owned by exactly one thread
    }
    __syncthreads();

    // ---- cell update + stats for output LayerNorm over tanh(cy) -----------
    float cyv[4], tcv[4], ogv[4];
    float tsum = 0.f, tsq = 0.f;
    #pragma unroll
    for (int r = 0; r < 4; ++r) {
        const int j  = t + (r << 8);
        const float ig = sI[j];
        const float fg = sI[j + HDIM];
        const float og = sI[j + 2 * HDIM];
        const float gg = sI[j + 3 * HDIM];
        const float c  = sigmoidf_fast(fg) * cx[(size_t)b * HDIM + j]
                       + sigmoidf_fast(ig) * tanhf(gg);
        const float tc = tanhf(c);
        cyv[r] = c; tcv[r] = tc; ogv[r] = og;
        tsum += tc; tsq += tc * tc;
    }
    float4 tt = block_reduce4(make_float4(tsum, tsq, 0.f, 0.f), red, t);
    const float invH = 1.0f / (float)HDIM;
    const float m2 = tt.x * invH;
    const float v2 = tt.y * invH - m2 * m2;
    const float r2 = rsqrtf(v2 + 1e-5f);

    #pragma unroll
    for (int r = 0; r < 4; ++r) {
        const int j = t + (r << 8);
        const float hv = sigmoidf_fast(ogv[r]) * ((tcv[r] - m2) * r2 * g_ho[j] + be_ho[j]);
        hy[(size_t)b * HDIM + j] = hv;
        cy[(size_t)b * HDIM + j] = cyv[r];
    }
}

// ---------------------------------------------------------------------------

extern "C" void kernel_launch(void* const* d_in, const int* in_sizes, int n_in,
                              void* d_out, int out_size, void* d_ws, size_t ws_size,
                              hipStream_t stream) {
    const float* inputs  = (const float*)d_in[0];
    const float* hx      = (const float*)d_in[1];
    const float* cx      = (const float*)d_in[2];
    const float* w_ih    = (const float*)d_in[3];
    const float* w_hh    = (const float*)d_in[4];
    const float* b_hh    = (const float*)d_in[5];
    const float* g_ih    = (const float*)d_in[6];
    const float* beta_ih = (const float*)d_in[7];
    const float* g_hh    = (const float*)d_in[8];
    const float* beta_hh = (const float*)d_in[9];
    const float* g_ho    = (const float*)d_in[10];
    const float* beta_ho = (const float*)d_in[11];

    float* i2h = (float*)d_ws;                        // 4096*4096 f32
    float* h2h = i2h + (size_t)MDIM * NDIM;           // 4096*4096 f32  (128 MB total)

    float* hy  = (float*)d_out;                       // tuple order: hy, cy
    float* cyo = hy + (size_t)MDIM * HDIM;

    dim3 g1(NDIM / BN, MDIM / BM, 2);
    ln_lstm_gemm<<<g1, 256, 0, stream>>>(inputs, w_ih, hx, w_hh, b_hh, i2h, h2h);
    ln_lstm_pointwise<<<MDIM, 256, 0, stream>>>(i2h, h2h, cx,
                                                g_ih, beta_ih, g_hh, beta_hh,
                                                g_ho, beta_ho, hy, cyo);
}
